// Net_45896020525237
// MI455X (gfx1250) — compile-verified, matrix operand order checked by reading, not run on hardware
//
#include <hip/hip_runtime.h>

// MLP: [B,4] -> 40 layers -> [B,1], f16 WMMA w/ f32 accumulate.
// Transposed recurrence h' = relu(W @ h^T): A-operand = W rows (A-format),
// B-operand = h^T (lane = batch row). The D layout of the 4 stacked 16x16
// f32 tiles is lane-compatible with the next layer's B operand, so repack
// is lane-local cvt_pk + v_pk_max -> zero cross-lane traffic in the loop.
// BT=4 batch tiles per wave amortizes per-layer LDS W-fragment reads so the
// WMMA pipe (not LDS @ ~256B/clk/WGP) is the limiter.

typedef __attribute__((ext_vector_type(16))) _Float16 v16h;
typedef __attribute__((ext_vector_type(8)))  float    v8f;

#define NH1 39            // 64x64 hidden layers
#define DH 64
#define THREADS 512       // 16 waves
#define BT 4              // batch tiles (16 rows) per wave
#define ROWS_PER_WG (THREADS / 32 * BT * 16)   // 1024

union V16H { v16h v; _Float16 h[16]; uint4 q[2]; };
union V8F  { v8f  v; float f[8];     float4 q[2]; };

__launch_bounds__(THREADS)
__global__ void mlp40_wmma(const float* __restrict__ x,
                           const float* __restrict__ Win,
                           const float* __restrict__ bin,
                           const float* __restrict__ Wh,
                           const float* __restrict__ bh,
                           const float* __restrict__ Wout,
                           const float* __restrict__ bout,
                           float* __restrict__ out)
{
    __shared__ __align__(16) _Float16 sW[2][DH * DH];   // double-buffered layer weights (f16)
    __shared__ __align__(16) float    sB[2][DH];        // double-buffered layer bias (f32)
    __shared__ __align__(16) float    sWin[DH * 4];
    __shared__ __align__(16) float    sBin[DH];
    __shared__ __align__(16) float    sWout[DH];

    const int tid  = threadIdx.x;
    const int lane = tid & 31;
    const int wid  = tid >> 5;
    const int lm   = lane & 15;   // batch row within tile / W row within j-tile
    const int hh   = lane >> 4;   // half selector for K/feature interleave

    // ---- stage first/last layer constants + hidden layer 0 weights (buf 0) ----
    if (tid < DH * 4)                 sWin[tid]        = Win[tid];
    if (tid >= 256 && tid < 256 + DH) sBin[tid - 256]  = bin[tid - 256];
    if (tid >= 320 && tid < 320 + DH) sWout[tid - 320] = Wout[tid - 320];
    {
        const float* src = Wh + tid * 8;            // layer 0: 4096 f32
        float4 a0 = *(const float4*)(src);
        float4 a1 = *(const float4*)(src + 4);
        __align__(16) _Float16 t8[8] = {
            (_Float16)a0.x, (_Float16)a0.y, (_Float16)a0.z, (_Float16)a0.w,
            (_Float16)a1.x, (_Float16)a1.y, (_Float16)a1.z, (_Float16)a1.w };
        *(uint4*)&sW[0][tid * 8] = *(const uint4*)t8;
        if (tid < DH) sB[0][tid] = bh[tid];
    }
    __syncthreads();

    // ---- layer 0: 4 -> 64 (f32 VALU), produce h in B-operand layout ----
    const long rowbase = (long)blockIdx.x * ROWS_PER_WG + (long)wid * (BT * 16);
    V16H hf[BT][2];  // [batch tile][K fragment]; lane holds its 32 features of its row
#pragma unroll
    for (int bt = 0; bt < BT; ++bt) {
        const long m = rowbase + bt * 16 + lm;
        float4 xv = *(const float4*)(x + m * 4);
#pragma unroll
        for (int f = 0; f < 2; ++f) {
#pragma unroll
            for (int e = 0; e < 16; ++e) {
                const int j = 32 * f + ((e >> 3) << 4) + 8 * hh + (e & 7);
                const float* wr = &sWin[j * 4];
                float v = sBin[j] + xv.x * wr[0] + xv.y * wr[1] + xv.z * wr[2] + xv.w * wr[3];
                hf[bt][f].h[e] = (_Float16)fmaxf(v, 0.0f);
            }
        }
    }

    const v16h hzero = {};

    // one hidden layer: 8 WMMAs per batch tile, weights/bias from sW/sB[buf]
    auto layer_compute = [&](int buf) {
        // bias as WMMA C operand: C[m=v+8h][n] = b[16t+8h+v], uniform over n
        V8F cin[4];
#pragma unroll
        for (int t = 0; t < 4; ++t) {
            cin[t].q[0] = *(const float4*)&sB[buf][16 * t + 8 * hh];
            cin[t].q[1] = *(const float4*)&sB[buf][16 * t + 8 * hh + 4];
        }
        // A operand: W rows. lane provides W[16t+lm][k], K interleaved by lane half.
        V16H wf[4][2];
#pragma unroll
        for (int t = 0; t < 4; ++t) {
            const _Float16* row = &sW[buf][(16 * t + lm) * DH];
#pragma unroll
            for (int f = 0; f < 2; ++f) {
                wf[t][f].q[0] = *(const uint4*)(row + 32 * f + 8 * hh);
                wf[t][f].q[1] = *(const uint4*)(row + 32 * f + 16 + 8 * hh);
            }
        }
#pragma unroll
        for (int bt = 0; bt < BT; ++bt) {
            v8f acc[4];
#pragma unroll
            for (int t = 0; t < 4; ++t)   // K = 0..31
                acc[t] = __builtin_amdgcn_wmma_f32_16x16x32_f16(
                    false, wf[t][0].v, false, hf[bt][0].v, (short)0, cin[t].v, false, false);
#pragma unroll
            for (int t = 0; t < 4; ++t)   // K = 32..63
                acc[t] = __builtin_amdgcn_wmma_f32_16x16x32_f16(
                    false, wf[t][1].v, false, hf[bt][1].v, (short)0, acc[t], false, false);

            // pack (v_cvt_pk_f16_f32) then ReLU on packed halves (v_pk_max_num_f16)
#pragma unroll
            for (int f = 0; f < 2; ++f) {
                V8F a0, a1; a0.v = acc[2 * f]; a1.v = acc[2 * f + 1];
#pragma unroll
                for (int e = 0; e < 8; ++e) hf[bt][f].h[e]     = (_Float16)a0.f[e];
#pragma unroll
                for (int e = 0; e < 8; ++e) hf[bt][f].h[e + 8] = (_Float16)a1.f[e];
                hf[bt][f].v = __builtin_elementwise_max(hf[bt][f].v, hzero);
            }
        }
    };

    // ---- hidden layers 0..37 with unconditional double-buffered prefetch ----
    for (int l = 0; l < NH1 - 1; ++l) {
        const int buf = l & 1;
        const float* src = Wh + (size_t)(l + 1) * DH * DH + tid * 8;
        float4 p0 = *(const float4*)(src);
        float4 p1 = *(const float4*)(src + 4);
        float  pb = bh[(l + 1) * DH + (tid & (DH - 1))];

        layer_compute(buf);

        __align__(16) _Float16 t8[8] = {
            (_Float16)p0.x, (_Float16)p0.y, (_Float16)p0.z, (_Float16)p0.w,
            (_Float16)p1.x, (_Float16)p1.y, (_Float16)p1.z, (_Float16)p1.w };
        *(uint4*)&sW[buf ^ 1][tid * 8] = *(const uint4*)t8;
        if (tid < DH) sB[buf ^ 1][tid] = pb;
        __syncthreads();
    }
    // ---- final hidden layer (l = 38, buf = 0), no prefetch ----
    layer_compute((NH1 - 1) & 1);

    // ---- output layer: 64 -> 1 ----
    const float bo = bout[0];
#pragma unroll
    for (int bt = 0; bt < BT; ++bt) {
        float p = 0.0f;
#pragma unroll
        for (int f = 0; f < 2; ++f) {
#pragma unroll
            for (int e = 0; e < 16; ++e) {
                const int j = 32 * f + ((e >> 3) << 4) + 8 * hh + (e & 7);
                p += (float)hf[bt][f].h[e] * sWout[j];
            }
        }
        p += __shfl_xor(p, 16, 32);   // combine the two feature halves (lane ^ 16)
        if (hh == 0) out[rowbase + bt * 16 + lm] = p + bo;
    }
}

extern "C" void kernel_launch(void* const* d_in, const int* in_sizes, int n_in,
                              void* d_out, int out_size, void* d_ws, size_t ws_size,
                              hipStream_t stream) {
    (void)n_in; (void)d_ws; (void)ws_size; (void)out_size;
    const float* x    = (const float*)d_in[0];
    const float* Win  = (const float*)d_in[1];
    const float* bin  = (const float*)d_in[2];
    const float* Wh   = (const float*)d_in[3];
    const float* bh   = (const float*)d_in[4];
    const float* Wout = (const float*)d_in[5];
    const float* bout = (const float*)d_in[6];
    float* out = (float*)d_out;

    const int B = in_sizes[0] / 4;             // x is [B,4]
    const int blocks = B / ROWS_PER_WG;        // 524288 / 1024 = 512 blocks
    mlp40_wmma<<<blocks, THREADS, 0, stream>>>(x, Win, bin, Wh, bh, Wout, bout, out);
}
